// ObjEdgeGaModule_11982958756644
// MI455X (gfx1250) — matrix operand order repaired and checked by reading, not compiled, not run on hardware
//
#include <hip/hip_runtime.h>
#include <hip/hip_bf16.h>

typedef __attribute__((ext_vector_type(16))) _Float16 v16h;
typedef __attribute__((ext_vector_type(8)))  _Float16 v8h;
typedef __attribute__((ext_vector_type(8)))  float    v8f;

#define DIMD 768
#define NHH  12
#define HD   64
#define NB   16
#define NOBJ 52
#define LL   24
#define DFFD 2048
#define EPSF 1e-5f

// ---------------------------------------------------------------------------
// gfx1250 async global->LDS DMA helpers (ASYNCcnt-tracked, no VGPR roundtrip).
// LDS operand is the low 32 bits of the generic address (ISA 10.2).
// 16-byte aligned b128 per lane.
// ---------------------------------------------------------------------------
__device__ __forceinline__ void async_copy_b128(const void* gptr, void* lptr) {
  unsigned lds = (unsigned)(size_t)lptr;
  unsigned long long ga = (unsigned long long)gptr;
  asm volatile("global_load_async_to_lds_b128 %0, %1, off"
               :: "v"(lds), "v"(ga) : "memory");
}
__device__ __forceinline__ void async_wait0() {
  asm volatile("s_wait_asynccnt 0x0" ::: "memory");
}

// ---------------------------------------------------------------------------
// Weight convert: W f32 [K,N] row-major -> Wt f16 [N,K] (so GEMM B tiles are
// contiguous along K and DMA into LDS with b128 transfers).
// ---------------------------------------------------------------------------
__global__ void k_convert_wT(const float* __restrict__ w, _Float16* __restrict__ wt,
                             int K, int N) {
  int idx = blockIdx.x * blockDim.x + threadIdx.x;
  if (idx >= K * N) return;
  int n = idx / K, k = idx - n * K;
  wt[idx] = (_Float16)w[(size_t)k * N + n];
}

// ---------------------------------------------------------------------------
// WMMA fragment helpers (CDNA5 16x16x32 f16 layouts, wave32):
//  A 16x32: lane<16 -> elems 0..7 = K0..7, elems 8..15 = K16..23 (row = lane&15)
//           lane>=16 -> K8..15 / K24..31
//  B 32x16: lane holds col = lane&15; same K order as A (K0..7,K16..23 / K8..15,K24..31)
//  C/D f32: elem r -> row r + (lane>=16 ? 8 : 0), col = lane&15
// ---------------------------------------------------------------------------
__device__ __forceinline__ v16h load_a_frag(const _Float16* rowp, int lane) {
  int hi = (lane >= 16) ? 8 : 0;
  v16h a;
  ((v8h*)&a)[0] = *(const v8h*)(rowp + hi);
  ((v8h*)&a)[1] = *(const v8h*)(rowp + 16 + hi);
  return a;
}
__device__ __forceinline__ v16h load_b_frag(const _Float16* colp, int lane) {
  int ko = (lane >= 16) ? 8 : 0;
  v16h b;
  ((v8h*)&b)[0] = *(const v8h*)(colp + ko);
  ((v8h*)&b)[1] = *(const v8h*)(colp + 16 + ko);
  return b;
}

// ---------------------------------------------------------------------------
// Generic tiled WMMA GEMM: Out[M,N] = A[M,K] * Bt[N,K]^T + bias (opt ReLU).
// Block = 256 threads (8 waves), tile 64x64, K chunked by 32, double-buffered
// LDS: next chunk staged (async DMA for f16 / convert for f32) while the
// current chunk's WMMAs run; one barrier per chunk.
// ---------------------------------------------------------------------------
template <typename AT, int OUTF16, int RELU>
__global__ __launch_bounds__(256) void k_gemm(const AT* __restrict__ A,
                                              const _Float16* __restrict__ Bt,
                                              const float* __restrict__ bias,
                                              void* __restrict__ Out,
                                              int M, int N, int K) {
  __shared__ __align__(16) _Float16 As[2][64][40];
  __shared__ __align__(16) _Float16 Bs[2][64][40];
  const int t = threadIdx.x, lane = t & 31, wave = t >> 5;
  const int wm = wave & 3, wn = wave >> 2;
  const int m0 = blockIdx.x * 64, n0 = blockIdx.y * 64;
  const int srow = t >> 2, sc8 = (t & 3) * 8;
  v8f acc0 = {}, acc1 = {};

  const AT* aRow = A + (size_t)(m0 + srow) * K + sc8;
  const _Float16* bRow = Bt + (size_t)(n0 + srow) * K + sc8;

  auto stage = [&](int buf, int kb) {
    async_copy_b128(bRow + kb, &Bs[buf][srow][sc8]);
    if (sizeof(AT) == 4) {             // f32 source: convert through VALU
      const AT* ap = aRow + kb;
      v8h v;
#pragma unroll
      for (int e = 0; e < 8; ++e) v[e] = (_Float16)ap[e];
      *(v8h*)&As[buf][srow][sc8] = v;
    } else {                           // f16 source: pure async DMA
      async_copy_b128(aRow + kb, &As[buf][srow][sc8]);
    }
  };

  stage(0, 0);
  int buf = 0;
  for (int kb = 0; kb < K; kb += 32) {
    async_wait0();
    __syncthreads();
    if (kb + 32 < K) stage(buf ^ 1, kb + 32);
    v16h af  = load_a_frag(&As[buf][wm * 16 + (lane & 15)][0], lane);
    v16h bf0 = load_b_frag(&Bs[buf][wn * 32 + (lane & 15)][0], lane);
    v16h bf1 = load_b_frag(&Bs[buf][wn * 32 + 16 + (lane & 15)][0], lane);
    acc0 = __builtin_amdgcn_wmma_f32_16x16x32_f16(false, af, false, bf0, (short)0, acc0, false, false);
    acc1 = __builtin_amdgcn_wmma_f32_16x16x32_f16(false, af, false, bf1, (short)0, acc1, false, false);
    buf ^= 1;
  }

  const int colA = n0 + wn * 32 + (lane & 15);
  const int colB = colA + 16;
  const int rbase = m0 + wm * 16 + ((lane >= 16) ? 8 : 0);
#pragma unroll
  for (int r = 0; r < 8; ++r) {
    int row = rbase + r;
    float v0 = acc0[r] + bias[colA];
    float v1 = acc1[r] + bias[colB];
    if (RELU) { v0 = fmaxf(v0, 0.f); v1 = fmaxf(v1, 0.f); }
    if (OUTF16) {
      _Float16* o = (_Float16*)Out;
      o[(size_t)row * N + colA] = (_Float16)v0;
      o[(size_t)row * N + colB] = (_Float16)v1;
    } else {
      float* o = (float*)Out;
      o[(size_t)row * N + colA] = v0;
      o[(size_t)row * N + colB] = v1;
    }
  }
}

// ---------------------------------------------------------------------------
// Edge-Q GEMM: A generated on the fly as diff[m,k] = obj[n,i,k]-obj[n,j,k],
// m = (n*52+i)*52+j.  Out f16 = diff @ e_wq + e_bq.  M=43264, N=K=768.
// B staged by async DMA; A produced in VALU (sub + cvt) under the WMMAs.
// ---------------------------------------------------------------------------
__global__ __launch_bounds__(256) void k_gemm_diff(const float* __restrict__ obj,
                                                   const _Float16* __restrict__ Bt,
                                                   const float* __restrict__ bias,
                                                   _Float16* __restrict__ Out,
                                                   int N, int K) {
  __shared__ __align__(16) _Float16 As[2][64][40];
  __shared__ __align__(16) _Float16 Bs[2][64][40];
  const int t = threadIdx.x, lane = t & 31, wave = t >> 5;
  const int wm = wave & 3, wn = wave >> 2;
  const int m0 = blockIdx.x * 64, n0 = blockIdx.y * 64;
  const int srow = t >> 2, sc8 = (t & 3) * 8;
  const int m = m0 + srow;
  const int nn = m / (NOBJ * NOBJ);
  const int rem = m % (NOBJ * NOBJ);
  const int oi = rem / NOBJ, oj = rem % NOBJ;
  const float* pib = obj + (size_t)(nn * NOBJ + oi) * K + sc8;
  const float* pjb = obj + (size_t)(nn * NOBJ + oj) * K + sc8;
  const _Float16* bRow = Bt + (size_t)(n0 + srow) * K + sc8;
  v8f acc0 = {}, acc1 = {};

  auto stage = [&](int buf, int kb) {
    async_copy_b128(bRow + kb, &Bs[buf][srow][sc8]);
    const float* pi = pib + kb;
    const float* pj = pjb + kb;
    v8h v;
#pragma unroll
    for (int e = 0; e < 8; ++e) v[e] = (_Float16)(pi[e] - pj[e]);
    *(v8h*)&As[buf][srow][sc8] = v;
  };

  stage(0, 0);
  int buf = 0;
  for (int kb = 0; kb < K; kb += 32) {
    async_wait0();
    __syncthreads();
    if (kb + 32 < K) stage(buf ^ 1, kb + 32);
    v16h af  = load_a_frag(&As[buf][wm * 16 + (lane & 15)][0], lane);
    v16h bf0 = load_b_frag(&Bs[buf][wn * 32 + (lane & 15)][0], lane);
    v16h bf1 = load_b_frag(&Bs[buf][wn * 32 + 16 + (lane & 15)][0], lane);
    acc0 = __builtin_amdgcn_wmma_f32_16x16x32_f16(false, af, false, bf0, (short)0, acc0, false, false);
    acc1 = __builtin_amdgcn_wmma_f32_16x16x32_f16(false, af, false, bf1, (short)0, acc1, false, false);
    buf ^= 1;
  }

  const int colA = n0 + wn * 32 + (lane & 15);
  const int colB = colA + 16;
  const int rbase = m0 + wm * 16 + ((lane >= 16) ? 8 : 0);
#pragma unroll
  for (int r = 0; r < 8; ++r) {
    int row = rbase + r;
    Out[(size_t)row * N + colA] = (_Float16)(acc0[r] + bias[colA]);
    Out[(size_t)row * N + colB] = (_Float16)(acc1[r] + bias[colB]);
  }
}

// ---------------------------------------------------------------------------
// Edge output-proj GEMM fused with max over partner j:
// block = (ni = n*52+i, column tile).  A rows = eatt[(ni*52+j), :]; padded
// rows j=52..63 just re-read row 51 (pure async DMA) -- they only affect
// output rows that the max-reduction excludes.  edge[ni,col] = max_j + e_bo.
// ---------------------------------------------------------------------------
__global__ __launch_bounds__(256) void k_gemm_maxfuse(const _Float16* __restrict__ A,
                                                      const _Float16* __restrict__ Bt,
                                                      const float* __restrict__ bias,
                                                      float* __restrict__ edge,
                                                      int K) {
  __shared__ __align__(16) _Float16 As[2][64][40];
  __shared__ __align__(16) _Float16 Bs[2][64][40];
  __shared__ float red[4][2][64];
  const int t = threadIdx.x, lane = t & 31, wave = t >> 5;
  const int wm = wave & 3, wn = wave >> 2;
  const int ni = blockIdx.x, n0 = blockIdx.y * 64;
  const int srow = t >> 2, sc8 = (t & 3) * 8;
  const int arow = (srow < NOBJ) ? srow : (NOBJ - 1);
  const _Float16* aRow = A + (size_t)(ni * NOBJ + arow) * K + sc8;
  const _Float16* bRow = Bt + (size_t)(n0 + srow) * K + sc8;
  v8f acc0 = {}, acc1 = {};

  auto stage = [&](int buf, int kb) {
    async_copy_b128(bRow + kb, &Bs[buf][srow][sc8]);
    async_copy_b128(aRow + kb, &As[buf][srow][sc8]);
  };

  stage(0, 0);
  int buf = 0;
  for (int kb = 0; kb < K; kb += 32) {
    async_wait0();
    __syncthreads();
    if (kb + 32 < K) stage(buf ^ 1, kb + 32);
    v16h af  = load_a_frag(&As[buf][wm * 16 + (lane & 15)][0], lane);
    v16h bf0 = load_b_frag(&Bs[buf][wn * 32 + (lane & 15)][0], lane);
    v16h bf1 = load_b_frag(&Bs[buf][wn * 32 + 16 + (lane & 15)][0], lane);
    acc0 = __builtin_amdgcn_wmma_f32_16x16x32_f16(false, af, false, bf0, (short)0, acc0, false, false);
    acc1 = __builtin_amdgcn_wmma_f32_16x16x32_f16(false, af, false, bf1, (short)0, acc1, false, false);
    buf ^= 1;
  }

  // per-lane masked max over the 8 accumulator rows (skip padded j >= 52)
  float mx0 = -3.4e38f, mx1 = -3.4e38f;
  const int rbase = wm * 16 + ((lane >= 16) ? 8 : 0);
#pragma unroll
  for (int r = 0; r < 8; ++r) {
    if (rbase + r < NOBJ) {
      mx0 = fmaxf(mx0, acc0[r]);
      mx1 = fmaxf(mx1, acc1[r]);
    }
  }
  const int half = lane >> 4;
  const int cl0 = wn * 32 + (lane & 15);
  red[wm][half][cl0] = mx0;
  red[wm][half][cl0 + 16] = mx1;
  __syncthreads();
  if (t < 64) {
    float mval = -3.4e38f;
#pragma unroll
    for (int a = 0; a < 4; ++a)
#pragma unroll
      for (int h = 0; h < 2; ++h) mval = fmaxf(mval, red[a][h][t]);
    edge[(size_t)ni * DIMD + n0 + t] = mval + bias[n0 + t];
  }
}

// ---------------------------------------------------------------------------
// Node attention: one warp per (n, h, q).  L=24 keys, softmax over lanes.
// ---------------------------------------------------------------------------
__global__ __launch_bounds__(256) void k_node_attn(const float* __restrict__ qh,
                                                   const float* __restrict__ kh,
                                                   const float* __restrict__ vh,
                                                   const int* __restrict__ co,
                                                   float* __restrict__ out) {
  const int wid = (int)((blockIdx.x * blockDim.x + threadIdx.x) >> 5);
  const int lane = threadIdx.x & 31;
  const int TOT = NB * NHH * NOBJ;
  if (wid >= TOT) return;
  const int n = wid / (NHH * NOBJ);
  const int rem = wid % (NHH * NOBJ);
  const int h = rem / NOBJ, q = rem % NOBJ;
  const float* qp = qh + (size_t)(n * NOBJ + q) * DIMD + h * HD;

  float s = -1e30f;
  if (lane < LL) {
    const float* kp = kh + (size_t)(n * LL + lane) * DIMD + h * HD;
    float a = 0.f;
#pragma unroll 8
    for (int d = 0; d < HD; ++d) a += qp[d] * kp[d];
    a *= 0.125f;
    s = (co[(size_t)n * LL * NOBJ + lane * NOBJ + q] > 0) ? a : -1e9f;
  }
  float mx = s;
  for (int o = 16; o > 0; o >>= 1) mx = fmaxf(mx, __shfl_xor(mx, o));
  float p = (lane < LL) ? __expf(s - mx) : 0.f;
  float sm = p;
  for (int o = 16; o > 0; o >>= 1) sm += __shfl_xor(sm, o);
  const float inv = 1.f / sm;
  float a0 = 0.f, a1 = 0.f;
  for (int l = 0; l < LL; ++l) {
    float pl = __shfl(p, l);
    const float* vp = vh + (size_t)(n * LL + l) * DIMD + h * HD;
    a0 += pl * vp[lane];
    a1 += pl * vp[lane + 32];
  }
  float* op = out + (size_t)(n * NOBJ + q) * DIMD + h * HD;
  op[lane] = a0 * inv;
  op[lane + 32] = a1 * inv;
}

// ---------------------------------------------------------------------------
// Edge attention: one warp per (n, h, i, j); q from f16, out f16.
// mask depends on (n, j, l) per reference broadcast.
// ---------------------------------------------------------------------------
__global__ __launch_bounds__(256) void k_edge_attn(const _Float16* __restrict__ qhe,
                                                   const float* __restrict__ kh,
                                                   const float* __restrict__ vh,
                                                   const int* __restrict__ co,
                                                   _Float16* __restrict__ out) {
  const int wid = (int)((blockIdx.x * blockDim.x + threadIdx.x) >> 5);
  const int lane = threadIdx.x & 31;
  const int TOT = NB * NHH * NOBJ * NOBJ;
  if (wid >= TOT) return;
  const int n = wid / (NHH * NOBJ * NOBJ);
  const int rem = wid % (NHH * NOBJ * NOBJ);
  const int h = rem / (NOBJ * NOBJ);
  const int ij = rem % (NOBJ * NOBJ);
  const int j = ij % NOBJ;
  const size_t m = (size_t)n * NOBJ * NOBJ + ij;
  const _Float16* qp = qhe + m * DIMD + h * HD;

  float s = -1e30f;
  if (lane < LL) {
    const float* kp = kh + (size_t)(n * LL + lane) * DIMD + h * HD;
    float a = 0.f;
#pragma unroll 8
    for (int d = 0; d < HD; ++d) a += (float)qp[d] * kp[d];
    a *= 0.125f;
    s = (co[(size_t)n * LL * NOBJ + lane * NOBJ + j] > 0) ? a : -1e9f;
  }
  float mx = s;
  for (int o = 16; o > 0; o >>= 1) mx = fmaxf(mx, __shfl_xor(mx, o));
  float p = (lane < LL) ? __expf(s - mx) : 0.f;
  float sm = p;
  for (int o = 16; o > 0; o >>= 1) sm += __shfl_xor(sm, o);
  const float inv = 1.f / sm;
  float a0 = 0.f, a1 = 0.f;
  for (int l = 0; l < LL; ++l) {
    float pl = __shfl(p, l);
    const float* vp = vh + (size_t)(n * LL + l) * DIMD + h * HD;
    a0 += pl * vp[lane];
    a1 += pl * vp[lane + 32];
  }
  _Float16* op = out + m * DIMD + h * HD;
  op[lane] = (_Float16)(a0 * inv);
  op[lane + 32] = (_Float16)(a1 * inv);
}

// ---------------------------------------------------------------------------
// LayerNorm over DIM=768 of (xa + xb); optional f32 output and/or f16 output
// written at [row*stride16 + coloff + c] (builds the concat FFN input).
// One block (256 thr) per row.
// ---------------------------------------------------------------------------
__global__ __launch_bounds__(256) void k_ln(const float* __restrict__ xa,
                                            const float* __restrict__ xb,
                                            const float* __restrict__ g,
                                            const float* __restrict__ b,
                                            float* __restrict__ of32,
                                            _Float16* __restrict__ of16,
                                            int coloff, int stride16) {
  __shared__ float red[16];
  const int row = blockIdx.x, t = threadIdx.x, w = t >> 5, lane = t & 31;
  const float* pa = xa + (size_t)row * DIMD;
  const float* pb = xb + (size_t)row * DIMD;
  float x[3];
  float s = 0.f;
#pragma unroll
  for (int e = 0; e < 3; ++e) {
    int c = t + e * 256;
    x[e] = pa[c] + pb[c];
    s += x[e];
  }
  for (int o = 16; o > 0; o >>= 1) s += __shfl_xor(s, o);
  if (lane == 0) red[w] = s;
  __syncthreads();
  float tot = 0.f;
#pragma unroll
  for (int i = 0; i < 8; ++i) tot += red[i];
  const float mean = tot * (1.f / DIMD);
  float vs = 0.f;
#pragma unroll
  for (int e = 0; e < 3; ++e) {
    float d = x[e] - mean;
    vs += d * d;
  }
  for (int o = 16; o > 0; o >>= 1) vs += __shfl_xor(vs, o);
  if (lane == 0) red[8 + w] = vs;
  __syncthreads();
  float var = 0.f;
#pragma unroll
  for (int i = 0; i < 8; ++i) var += red[8 + i];
  const float rstd = rsqrtf(var * (1.f / DIMD) + EPSF);
#pragma unroll
  for (int e = 0; e < 3; ++e) {
    int c = t + e * 256;
    float y = (x[e] - mean) * rstd * g[c] + b[c];
    if (of32) of32[(size_t)row * DIMD + c] = y;
    if (of16) of16[(size_t)row * stride16 + coloff + c] = (_Float16)y;
  }
}

// ---------------------------------------------------------------------------
extern "C" void kernel_launch(void* const* d_in, const int* in_sizes, int n_in,
                              void* d_out, int out_size, void* d_ws, size_t ws_size,
                              hipStream_t stream) {
  (void)in_sizes; (void)n_in; (void)out_size; (void)ws_size;
  const float* lang = (const float*)d_in[0];
  const float* obj  = (const float*)d_in[1];
  const int*   co   = (const int*)d_in[4];
  const float* e_wq = (const float*)d_in[5];  const float* e_bq = (const float*)d_in[6];
  const float* e_wk = (const float*)d_in[7];  const float* e_bk = (const float*)d_in[8];
  const float* e_wv = (const float*)d_in[9];  const float* e_bv = (const float*)d_in[10];
  const float* e_wo = (const float*)d_in[11]; const float* e_bo = (const float*)d_in[12];
  const float* n_wq = (const float*)d_in[13]; const float* n_bq = (const float*)d_in[14];
  const float* n_wk = (const float*)d_in[15]; const float* n_bk = (const float*)d_in[16];
  const float* n_wv = (const float*)d_in[17]; const float* n_bv = (const float*)d_in[18];
  const float* n_wo = (const float*)d_in[19]; const float* n_bo = (const float*)d_in[20];
  const float* fw1  = (const float*)d_in[21]; const float* fb1  = (const float*)d_in[22];
  const float* fw2  = (const float*)d_in[23]; const float* fb2  = (const float*)d_in[24];
  const float* lne_g = (const float*)d_in[25]; const float* lne_b = (const float*)d_in[26];
  const float* lnq_g = (const float*)d_in[27]; const float* lnq_b = (const float*)d_in[28];
  const float* ln2_g = (const float*)d_in[29]; const float* ln2_b = (const float*)d_in[30];

  const int MO = NB * NOBJ;            // 832
  const int ML = NB * LL;              // 384
  const int ME = NB * NOBJ * NOBJ;     // 43264

  char* p = (char*)d_ws;
  auto carve = [&](size_t bytes) -> char* {
    char* r = p;
    p += (bytes + 255) & ~(size_t)255;
    return r;
  };
  _Float16* t_ewq = (_Float16*)carve((size_t)DIMD * DIMD * 2);
  _Float16* t_ewk = (_Float16*)carve((size_t)DIMD * DIMD * 2);
  _Float16* t_ewv = (_Float16*)carve((size_t)DIMD * DIMD * 2);
  _Float16* t_ewo = (_Float16*)carve((size_t)DIMD * DIMD * 2);
  _Float16* t_nwq = (_Float16*)carve((size_t)DIMD * DIMD * 2);
  _Float16* t_nwk = (_Float16*)carve((size_t)DIMD * DIMD * 2);
  _Float16* t_nwv = (_Float16*)carve((size_t)DIMD * DIMD * 2);
  _Float16* t_nwo = (_Float16*)carve((size_t)DIMD * DIMD * 2);
  _Float16* t_fw1 = (_Float16*)carve((size_t)2 * DIMD * DFFD * 2);
  _Float16* t_fw2 = (_Float16*)carve((size_t)DFFD * DIMD * 2);
  float* qh       = (float*)carve((size_t)MO * DIMD * 4);
  float* kh       = (float*)carve((size_t)ML * DIMD * 4);
  float* vh       = (float*)carve((size_t)ML * DIMD * 4);
  float* khe      = (float*)carve((size_t)ML * DIMD * 4);
  float* vhe      = (float*)carve((size_t)ML * DIMD * 4);
  _Float16* qhe   = (_Float16*)carve((size_t)ME * DIMD * 2);
  _Float16* eatt  = (_Float16*)carve((size_t)ME * DIMD * 2);
  float* natt     = (float*)carve((size_t)MO * DIMD * 4);
  float* node_o   = (float*)carve((size_t)MO * DIMD * 4);
  float* edge     = (float*)carve((size_t)MO * DIMD * 4);
  float* xn       = (float*)carve((size_t)MO * DIMD * 4);
  _Float16* cat16 = (_Float16*)carve((size_t)MO * 2 * DIMD * 2);
  _Float16* fmid  = (_Float16*)carve((size_t)MO * DFFD * 2);
  float* fout     = (float*)carve((size_t)MO * DIMD * 4);

  // --- 1. weight conversion ---
  {
    int el = DIMD * DIMD, blk = (el + 255) / 256;
    k_convert_wT<<<blk, 256, 0, stream>>>(e_wq, t_ewq, DIMD, DIMD);
    k_convert_wT<<<blk, 256, 0, stream>>>(e_wk, t_ewk, DIMD, DIMD);
    k_convert_wT<<<blk, 256, 0, stream>>>(e_wv, t_ewv, DIMD, DIMD);
    k_convert_wT<<<blk, 256, 0, stream>>>(e_wo, t_ewo, DIMD, DIMD);
    k_convert_wT<<<blk, 256, 0, stream>>>(n_wq, t_nwq, DIMD, DIMD);
    k_convert_wT<<<blk, 256, 0, stream>>>(n_wk, t_nwk, DIMD, DIMD);
    k_convert_wT<<<blk, 256, 0, stream>>>(n_wv, t_nwv, DIMD, DIMD);
    k_convert_wT<<<blk, 256, 0, stream>>>(n_wo, t_nwo, DIMD, DIMD);
    int el1 = 2 * DIMD * DFFD;
    k_convert_wT<<<(el1 + 255) / 256, 256, 0, stream>>>(fw1, t_fw1, 2 * DIMD, DFFD);
    int el2 = DFFD * DIMD;
    k_convert_wT<<<(el2 + 255) / 256, 256, 0, stream>>>(fw2, t_fw2, DFFD, DIMD);
  }

  // --- 2. projections (WMMA GEMMs) ---
  dim3 gL(ML / 64, DIMD / 64);   // 6 x 12
  dim3 gO(MO / 64, DIMD / 64);   // 13 x 12
  k_gemm<float, 0, 0><<<gL, 256, 0, stream>>>(lang, t_nwk, n_bk, kh,  ML, DIMD, DIMD);
  k_gemm<float, 0, 0><<<gL, 256, 0, stream>>>(lang, t_nwv, n_bv, vh,  ML, DIMD, DIMD);
  k_gemm<float, 0, 0><<<gL, 256, 0, stream>>>(lang, t_ewk, e_bk, khe, ML, DIMD, DIMD);
  k_gemm<float, 0, 0><<<gL, 256, 0, stream>>>(lang, t_ewv, e_bv, vhe, ML, DIMD, DIMD);
  k_gemm<float, 0, 0><<<gO, 256, 0, stream>>>(obj,  t_nwq, n_bq, qh,  MO, DIMD, DIMD);

  // --- 3. edge Q: diff @ e_wq + e_bq (A generated on the fly) ---
  dim3 gE(ME / 64, DIMD / 64);   // 676 x 12
  k_gemm_diff<<<gE, 256, 0, stream>>>(obj, t_ewq, e_bq, qhe, DIMD, DIMD);

  // --- 4. node attention ---
  {
    int warps = NB * NHH * NOBJ;             // 9984
    k_node_attn<<<(warps + 7) / 8, 256, 0, stream>>>(qh, kh, vh, co, natt);
  }
  // node output projection
  k_gemm<float, 0, 0><<<gO, 256, 0, stream>>>(natt, t_nwo, n_bo, node_o, MO, DIMD, DIMD);

  // --- 5. edge attention ---
  {
    int warps = NB * NHH * NOBJ * NOBJ;      // 519168
    k_edge_attn<<<(warps + 7) / 8, 256, 0, stream>>>(qhe, khe, vhe, co, eatt);
  }
  // edge output projection fused with max over j
  {
    dim3 gM(MO, DIMD / 64);                  // 832 x 12
    k_gemm_maxfuse<<<gM, 256, 0, stream>>>(eatt, t_ewo, e_bo, edge, DIMD);
  }

  // --- 6. fuse: xn = LN(obj+node), xe = LN(obj+edge); build concat f16 ---
  k_ln<<<MO, 256, 0, stream>>>(obj, node_o, lnq_g, lnq_b, xn, cat16, 0, 2 * DIMD);
  k_ln<<<MO, 256, 0, stream>>>(obj, edge,   lne_g, lne_b, (float*)nullptr, cat16, DIMD, 2 * DIMD);

  // --- 7. FFN ---
  dim3 gF1(MO / 64, DFFD / 64);  // 13 x 32
  k_gemm<_Float16, 1, 1><<<gF1, 256, 0, stream>>>(cat16, t_fw1, fb1, fmid, MO, DFFD, 2 * DIMD);
  dim3 gF2(MO / 64, DIMD / 64);  // 13 x 12
  k_gemm<_Float16, 0, 0><<<gF2, 256, 0, stream>>>(fmid, t_fw2, fb2, fout, MO, DIMD, DFFD);

  // --- 8. final LN -> d_out ---
  k_ln<<<MO, 256, 0, stream>>>(xn, fout, ln2_g, ln2_b, (float*)d_out, (_Float16*)nullptr, 0, 0);
}
